// Chuncked_Self_Attn_FM_22222160789941
// MI455X (gfx1250) — matrix operand order repaired and checked by reading, not compiled, not run on hardware
//
#include <hip/hip_runtime.h>
#include <hip/hip_bf16.h>

// ---------------------------------------------------------------------------
// Chunked self-attention feature map for MI455X (gfx1250), wave32 + WMMA bf16.
// 1 block = 1 chunk (512 blocks), 256 threads = 8 waves.
// v3: cap unrolling of the stage-1 channel-tile loop (and outer tile loops)
//     to eliminate the scratch spills/reloads seen in v2; keeps direct
//     global->register B fragments and pipelined LDS fragment loads.
// ---------------------------------------------------------------------------

typedef __attribute__((ext_vector_type(16))) __bf16 bf16x16;
typedef __attribute__((ext_vector_type(8)))  __bf16 bf16x8;
typedef __attribute__((ext_vector_type(8)))  float  f32x8;

#define QT_STR  40      // bf16 per qT/aoT row (32 + pad) -> 80B = 5*16B aligned
#define KP_STR  40      // bf16 per pooled-K row
#define VP_STR  264     // bf16 per pooled-V row (256 + pad) -> 528B = 33*16B
#define STG_STR 264     // per-wave attn staging row stride

#define QT_OFF   0
#define QT_BYTES (1024 * QT_STR * 2)            // 81920
#define KP_OFF   (QT_OFF + QT_BYTES)
#define KP_BYTES (256 * KP_STR * 2)             // 20480
#define VP_OFF   (KP_OFF + KP_BYTES)
#define VP_BYTES (32 * VP_STR * 2)              // 16896
#define STG_OFF  (VP_OFF + VP_BYTES)
#define STG_BYTES (8 * 16 * STG_STR * 2)        // 67584
#define SMEM_BYTES (STG_OFF + STG_BYTES)        // 186880 (< 320KB WGP LDS)

__device__ __forceinline__ f32x8 wmma_bf16(bf16x16 a, bf16x16 b, f32x8 c) {
  return __builtin_amdgcn_wmma_f32_16x16x32_bf16(
      /*neg_a=*/false, a, /*neg_b=*/false, b,
      /*c_mod=*/(short)0, c, /*reuse_a=*/false, /*reuse_b=*/false);
}

// Build a 16-element bf16 fragment from two 16-byte LDS chunks.
__device__ __forceinline__ bf16x16 frag2(const __bf16* p0, const __bf16* p1) {
  bf16x8 a = *(const bf16x8*)p0;
  bf16x8 b = *(const bf16x8*)p1;
  bf16x16 r;
#pragma unroll
  for (int i = 0; i < 8; ++i) { r[i] = a[i]; r[8 + i] = b[i]; }
  return r;
}

// A-operand fragment from a global fp32 row (W matrices), per-lane col base k0.
__device__ __forceinline__ bf16x16 a_frag_global(const float* row, int k0) {
  bf16x16 r;
#pragma unroll
  for (int i = 0; i < 8; ++i) {
    r[i]     = (__bf16)row[k0 + i];
    r[8 + i] = (__bf16)row[k0 + 16 + i];
  }
  return r;
}

// B-operand fragment of X directly from global: lane owns one pixel column
// (pcol), elements run over input channels (stride H*W = 65536 floats).
__device__ __forceinline__ bf16x16 b_frag_x(const float* pcol, int kbase) {
  bf16x16 r;
#pragma unroll
  for (int e = 0; e < 16; ++e)
    r[e] = (__bf16)pcol[(size_t)(kbase + e + (e >= 8 ? 8 : 0)) * 65536u];
  return r;
}

// Energy B fragment from pooled-K LDS (pix-major rows).
__device__ __forceinline__ bf16x16 kp_bfrag(const __bf16* KpT, int kt, int lo, int hi) {
  const __bf16* p = KpT + (kt * 16 + lo) * KP_STR + (hi ? 8 : 0);
  return frag2(p, p + 16);
}

__global__ __launch_bounds__(256, 1) void chunk_attn_kernel(
    const float* __restrict__ x,
    const float* __restrict__ Wq, const float* __restrict__ bq,
    const float* __restrict__ Wk, const float* __restrict__ bk,
    const float* __restrict__ Wv, const float* __restrict__ bv,
    const float* __restrict__ Wo, const float* __restrict__ bo,
    const float* __restrict__ gamma,
    float* __restrict__ out)
{
  extern __shared__ char smem[];
  __bf16* qT  = (__bf16*)(smem + QT_OFF);   // [1024][QT_STR]  q^T, later ao^T
  __bf16* KpT = (__bf16*)(smem + KP_OFF);   // [256][KP_STR]   pooled K, pix-major
  __bf16* Vp  = (__bf16*)(smem + VP_OFF);   // [32][VP_STR]    pooled V, ch-major
  __bf16* stg = (__bf16*)(smem + STG_OFF);  // [8][16][STG_STR] per-wave attn

  const int tid  = threadIdx.x;
  const int wv   = tid >> 5;
  const int lane = tid & 31;
  const int lo   = lane & 15;
  const int hi   = lane >> 4;        // wave half

  const int blk = blockIdx.x;        // 0..511
  const int n   = blk >> 6;
  const int gi  = (blk >> 3) & 7;
  const int gj  = blk & 7;
  const size_t chunk_base =
      (size_t)n * 256u * 65536u + (size_t)(gi * 32) * 256u + (size_t)(gj * 32);

  __bf16* stg_w = stg + wv * (16 * STG_STR);

  // Pixel tile = 2 rows x 8 cols spatial block; tile id t -> (rpair=t/4, wb=t%4)
  // in-tile pixel m: h = 2*rpair + (m>>3), w = wb*8 + (m&7).
  const int mh = lo >> 3;            // in-tile h of this lane's N column
  const int mw = lo & 7;

  // ============================ Stage 1: QKV =============================
#pragma unroll 1
  for (int it = 0; it < 8; ++it) {
    const int pt = it * 8 + wv;      // 0..63
    const int rp = pt >> 2;
    const int wb = pt & 3;

    // This lane's pixel column base in global x.
    const float* pcol = x + chunk_base + (size_t)(2 * rp + mh) * 256u
                          + (size_t)(wb * 8 + mw);

    // Load all 8 K-step B fragments of X into registers (reused by q,k,v).
    const int b_koff = hi ? 8 : 0;
    bf16x16 bx[8];
#pragma unroll
    for (int ks = 0; ks < 8; ++ks)
      bx[ks] = b_frag_x(pcol, ks * 32 + b_koff);

    const int a_off = hi ? 8 : 0;
    const int pixN  = (2 * rp + mh) * 32 + wb * 8 + mw;

    // NOTE: unroll 1 — keeps only one channel tile's A fragments live at a
    // time (bx 64 + A 64 + acc 8 VGPRs) to avoid scratch spills.
#pragma unroll 1
    for (int ct = 0; ct < 6; ++ct) {  // 0-1: q, 2-3: k, 4-5: v (16 ch each)
      const float* Wsel = (ct < 2) ? Wq : ((ct < 4) ? Wk : Wv);
      const float* bsel = (ct < 2) ? bq : ((ct < 4) ? bk : bv);
      const int chb = (ct & 1) * 16;
      const float* arow = Wsel + (size_t)(chb + lo) * 256u;

      f32x8 acc = {};
#pragma unroll
      for (int ks = 0; ks < 8; ++ks) {
        bf16x16 a = a_frag_global(arow, ks * 32 + a_off);
        acc = wmma_bf16(a, bx[ks], acc);
      }
#pragma unroll
      for (int q8 = 0; q8 < 8; ++q8) acc[q8] += bsel[chb + q8 + 8 * hi];

      if (ct < 2) {                  // q: write transposed to LDS
#pragma unroll
        for (int q8 = 0; q8 < 8; ++q8)
          qT[pixN * QT_STR + (ct * 16 + q8 + 8 * hi)] = (__bf16)acc[q8];
      } else {                       // k/v: 2x2 maxpool in-wave, then LDS
#pragma unroll
        for (int q8 = 0; q8 < 8; ++q8) {
          float v = acc[q8];
          v = fmaxf(v, __shfl_xor(v, 8));   // pool h pair (rows of tile)
          v = fmaxf(v, __shfl_xor(v, 1));   // pool w pair
          if ((lo < 8) && ((lo & 1) == 0)) {
            int kpix = rp * 16 + wb * 4 + (lo >> 1);   // pooled 16x16 grid
            int chn  = (ct & 1) * 16 + q8 + 8 * hi;
            if (ct < 4) KpT[kpix * KP_STR + chn] = (__bf16)v;
            else        Vp[chn * VP_STR + kpix]  = (__bf16)v;
          }
        }
      }
    }
  }
  __syncthreads();

  // ================= Stage 2: energy -> softmax -> ao^T ==================
#pragma unroll 1
  for (int it = 0; it < 8; ++it) {
    const int qt = it * 8 + wv;
    const int rp = qt >> 2;
    const int wb = qt & 3;

    // A = q^T rows of this tile [16 q, 32 c], single K=32 fragment.
    const int pixm = (2 * rp + mh) * 32 + wb * 8 + mw;
    const __bf16* qrow = qT + pixm * QT_STR + (hi ? 8 : 0);
    bf16x16 aq = frag2(qrow, qrow + 16);

    // energy [16 q, 256 k]: 16 WMMAs, B fragments double-buffered in groups
    // of 4 so ds_load latency overlaps the matrix pipe.
    f32x8 e[16];
    bf16x16 bcur[4], bnxt[4];
#pragma unroll
    for (int j = 0; j < 4; ++j) bcur[j] = kp_bfrag(KpT, j, lo, hi);
#pragma unroll
    for (int g = 0; g < 4; ++g) {
#pragma unroll
      for (int j = 0; j < 4; ++j)
        if (g < 3) bnxt[j] = kp_bfrag(KpT, (g + 1) * 4 + j, lo, hi);
#pragma unroll
      for (int j = 0; j < 4; ++j) {
        f32x8 z = {};
        e[g * 4 + j] = wmma_bf16(aq, bcur[j], z);
      }
#pragma unroll
      for (int j = 0; j < 4; ++j)
        if (g < 3) bcur[j] = bnxt[j];
    }

    // Softmax over 256 (16 kt in-lane x 16 lanes within half-wave).
#pragma unroll
    for (int q8 = 0; q8 < 8; ++q8) {
      float m0 = e[0][q8];
#pragma unroll
      for (int kt = 1; kt < 16; ++kt) m0 = fmaxf(m0, e[kt][q8]);
#pragma unroll
      for (int off = 1; off < 16; off <<= 1) m0 = fmaxf(m0, __shfl_xor(m0, off));
      float s0 = 0.f;
#pragma unroll
      for (int kt = 0; kt < 16; ++kt) {
        float p = __expf(e[kt][q8] - m0);
        e[kt][q8] = p;
        s0 += p;
      }
#pragma unroll
      for (int off = 1; off < 16; off <<= 1) s0 += __shfl_xor(s0, off);
      float inv = 1.0f / s0;
#pragma unroll
      for (int kt = 0; kt < 16; ++kt) e[kt][q8] *= inv;
    }

    // Stage attn tile bf16 into wave-private LDS (A-operand roundtrip).
#pragma unroll
    for (int kt = 0; kt < 16; ++kt)
#pragma unroll
      for (int q8 = 0; q8 < 8; ++q8)
        stg_w[(q8 + 8 * hi) * STG_STR + kt * 16 + lo] = (__bf16)e[kt][q8];

    // ao^T [16 q, 32 c] = attn [16,256] x v^T [256,32]; pipeline frag loads
    // one K-step ahead.
    f32x8 ao0 = {}, ao1 = {};
    bf16x16 a_c, b0_c, b1_c;
    {
      const __bf16* ar = stg_w + lo * STG_STR + (hi ? 8 : 0);
      a_c = frag2(ar, ar + 16);
      const __bf16* v0 = Vp + lo * VP_STR + (hi ? 8 : 0);
      b0_c = frag2(v0, v0 + 16);
      const __bf16* v1 = Vp + (16 + lo) * VP_STR + (hi ? 8 : 0);
      b1_c = frag2(v1, v1 + 16);
    }
#pragma unroll
    for (int ks = 0; ks < 8; ++ks) {
      bf16x16 a_n, b0_n, b1_n;
      if (ks < 7) {
        const __bf16* ar = stg_w + lo * STG_STR + (ks + 1) * 32 + (hi ? 8 : 0);
        a_n = frag2(ar, ar + 16);
        const __bf16* v0 = Vp + lo * VP_STR + (ks + 1) * 32 + (hi ? 8 : 0);
        b0_n = frag2(v0, v0 + 16);
        const __bf16* v1 = Vp + (16 + lo) * VP_STR + (ks + 1) * 32 + (hi ? 8 : 0);
        b1_n = frag2(v1, v1 + 16);
      }
      ao0 = wmma_bf16(a_c, b0_c, ao0);
      ao1 = wmma_bf16(a_c, b1_c, ao1);
      if (ks < 7) { a_c = a_n; b0_c = b0_n; b1_c = b1_n; }
    }
    // Overwrite consumed q^T rows with ao^T (same tile rows only).
#pragma unroll
    for (int q8 = 0; q8 < 8; ++q8) {
      int pm  = q8 + 8 * hi;
      int pix = (2 * rp + (pm >> 3)) * 32 + wb * 8 + (pm & 7);
      qT[pix * QT_STR + lo]      = (__bf16)ao0[q8];
      qT[pix * QT_STR + 16 + lo] = (__bf16)ao1[q8];
    }
  }
  __syncthreads();

  // ============ Stage 3: out = gamma*(Wo x ao + bo) + x ==================
  const float g = gamma[0];
#pragma unroll 1
  for (int mi = 0; mi < 2; ++mi) {
    const int mt = wv * 2 + mi;      // 16 output-channel tiles over 8 waves
    const float* arow = Wo + (size_t)(mt * 16 + lo) * 32u;
    bf16x16 a = a_frag_global(arow, hi ? 8 : 0);
    float bov[8];
#pragma unroll
    for (int q8 = 0; q8 < 8; ++q8) bov[q8] = bo[mt * 16 + q8 + 8 * hi];

    // Double-buffer the ao B fragment across the 64 pixel tiles.
    bf16x16 bcur;
    {
      const __bf16* br = qT + ((0 + mh) * 32 + 0 * 8 + mw) * QT_STR + (hi ? 8 : 0);
      bcur = frag2(br, br + 16);
    }
#pragma unroll 1
    for (int nt = 0; nt < 64; ++nt) {
      int rt = nt >> 2, wbt = nt & 3;
      bf16x16 bnxt;
      if (nt < 63) {
        int rn = (nt + 1) >> 2, wbn = (nt + 1) & 3;
        int pixn = (2 * rn + mh) * 32 + wbn * 8 + mw;
        const __bf16* br = qT + pixn * QT_STR + (hi ? 8 : 0);
        bnxt = frag2(br, br + 16);
      }
      f32x8 z = {};
      f32x8 acc = wmma_bf16(a, bcur, z);
#pragma unroll
      for (int q8 = 0; q8 < 8; ++q8) {
        int chn = mt * 16 + q8 + 8 * hi;
        size_t ga = chunk_base + (size_t)chn * 65536u
                  + (size_t)(2 * rt + mh) * 256u + (size_t)(wbt * 8 + mw);
        out[ga] = g * (acc[q8] + bov[q8]) + x[ga];
      }
      if (nt < 63) bcur = bnxt;
    }
  }
}

extern "C" void kernel_launch(void* const* d_in, const int* in_sizes, int n_in,
                              void* d_out, int out_size, void* d_ws, size_t ws_size,
                              hipStream_t stream) {
  (void)in_sizes; (void)n_in; (void)out_size; (void)d_ws; (void)ws_size;
  const float* x     = (const float*)d_in[0];
  const float* Wq    = (const float*)d_in[1];
  const float* bq    = (const float*)d_in[2];
  const float* Wk    = (const float*)d_in[3];
  const float* bk    = (const float*)d_in[4];
  const float* Wv    = (const float*)d_in[5];
  const float* bv    = (const float*)d_in[6];
  const float* Wo    = (const float*)d_in[7];
  const float* bo    = (const float*)d_in[8];
  const float* gamma = (const float*)d_in[9];
  float* out = (float*)d_out;

  dim3 grid(512), block(256);
  hipLaunchKernelGGL(chunk_attn_kernel, grid, block, SMEM_BYTES, stream,
                     x, Wq, bq, Wk, bk, Wv, bv, Wo, bo, gamma, out);
}
